// CyberSecLLMModel_31499290149093
// MI455X (gfx1250) — matrix operand order repaired and checked by reading, not compiled, not run on hardware
//
#include <hip/hip_runtime.h>
#include <hip/hip_bf16.h>
#include <math.h>

typedef __attribute__((ext_vector_type(16))) _Float16 v16h;
typedef __attribute__((ext_vector_type(8)))  float    v8f;

#define BATCH 16384
#define DIM   256
#define FFD   512
#define NEXP  8
#define KBR   32
#define NCLS  34

#define BM 64
#define BN 64
#define BK 32
#define BMF 128                     // fast-path M tile (8 waves)
#define AS_STRIDE (BK + 8)
#define BS_STRIDE (BK + 8)

#define ACT_NONE     0
#define ACT_GELU     1
#define ACT_SIGMOID  2
#define ACT_SILU     3
#define ACT_SOFTPLUS 4

template <int ACT>
__device__ __forceinline__ float act_apply(float x) {
  if (ACT == ACT_GELU)     return 0.5f * x * (1.0f + erff(x * 0.70710678118654752f));
  if (ACT == ACT_SIGMOID)  return 1.0f / (1.0f + expf(-x));
  if (ACT == ACT_SILU)     return x / (1.0f + expf(-x));
  if (ACT == ACT_SOFTPLUS) return (x > 20.0f) ? x : log1pf(expf(x));
  return x;
}

// Per-wave WMMA fragment gathers (16x16x32 f16; A and B both use k = 16*(lane/16) + s).
__device__ __forceinline__ v16h load_afrag(const _Float16* As, int wave, int lane) {
  const int ar  = (wave << 4) + (lane & 15);
  const int kb0 = (lane >> 4) << 4;
  v16h a;
#pragma unroll
  for (int i = 0; i < 4; i++) {
    a[2 * i]         = As[ar * AS_STRIDE + kb0 + 2 * i];
    a[2 * i + 1]     = As[ar * AS_STRIDE + kb0 + 2 * i + 1];
    a[8 + 2 * i]     = As[ar * AS_STRIDE + 8 + kb0 + 2 * i];
    a[8 + 2 * i + 1] = As[ar * AS_STRIDE + 8 + kb0 + 2 * i + 1];
  }
  return a;
}

__device__ __forceinline__ v16h load_bfrag(const _Float16* Bs, int t, int lane) {
  const int bn = (t << 4) + (lane & 15);
  const int kb = (lane >> 4) << 4;
  v16h b;
#pragma unroll
  for (int i = 0; i < 8; i++) {
    b[2 * i]     = Bs[bn * BS_STRIDE + kb + 2 * i];
    b[2 * i + 1] = Bs[bn * BS_STRIDE + kb + 2 * i + 1];
  }
  return b;
}

// ---------------------------------------------------------------------------
// FAST GEMM (M%128==0, N%64==0, K%32==0, lda/ldb%4==0).
// 256 threads = 8 waves; tile 128x64x32; float4 loads; register-staged
// software pipeline (next tile loads overlap current tile's WMMAs).
// ---------------------------------------------------------------------------
template <int ACT>
__global__ __launch_bounds__(256) void gemm_wmma_fast_kernel(
    const float* __restrict__ A, int lda,
    const float* __restrict__ B, int ldb,
    const float* __restrict__ bias,
    float* __restrict__ C, int ldc,
    int M, int N, int K)
{
  __shared__ _Float16 As[BMF * AS_STRIDE];     // [m][k]
  __shared__ _Float16 Bs[BN  * BS_STRIDE];     // transposed [n][k]

  const int tid  = threadIdx.x;
  const int wave = tid >> 5;
  const int lane = tid & 31;
  const int m0 = blockIdx.y * BMF;
  const int n0 = blockIdx.x * BN;

  // Per-thread staging coordinates
  const int ar4[4] = { (tid + 0)   >> 3, (tid + 256) >> 3,
                       (tid + 512) >> 3, (tid + 768) >> 3 };
  const int ac4    = (tid & 7) << 2;
  const int bk2[2] = { (tid + 0) >> 4, (tid + 256) >> 4 };
  const int bn4    = (tid & 15) << 2;

  v8f vzero = {};
  v8f acc[4];
#pragma unroll
  for (int t = 0; t < 4; t++) acc[t] = vzero;

  const int nk = K / BK;
  float4 ra[4], rb[2];
#pragma unroll
  for (int i = 0; i < 4; i++)
    ra[i] = *reinterpret_cast<const float4*>(A + (size_t)(m0 + ar4[i]) * lda + ac4);
#pragma unroll
  for (int i = 0; i < 2; i++)
    rb[i] = *reinterpret_cast<const float4*>(B + (size_t)bk2[i] * ldb + n0 + bn4);

  for (int kt = 0; kt < nk; kt++) {
    // Stage current tile into LDS (fp32 -> fp16)
#pragma unroll
    for (int i = 0; i < 4; i++) {
      _Float16* dst = &As[ar4[i] * AS_STRIDE + ac4];
      dst[0] = (_Float16)ra[i].x; dst[1] = (_Float16)ra[i].y;
      dst[2] = (_Float16)ra[i].z; dst[3] = (_Float16)ra[i].w;
    }
#pragma unroll
    for (int i = 0; i < 2; i++) {
      Bs[(bn4 + 0) * BS_STRIDE + bk2[i]] = (_Float16)rb[i].x;
      Bs[(bn4 + 1) * BS_STRIDE + bk2[i]] = (_Float16)rb[i].y;
      Bs[(bn4 + 2) * BS_STRIDE + bk2[i]] = (_Float16)rb[i].z;
      Bs[(bn4 + 3) * BS_STRIDE + bk2[i]] = (_Float16)rb[i].w;
    }
    __syncthreads();

    // Prefetch next tile (overlaps with WMMA compute below)
    if (kt + 1 < nk) {
      int k0 = (kt + 1) * BK;
#pragma unroll
      for (int i = 0; i < 4; i++)
        ra[i] = *reinterpret_cast<const float4*>(A + (size_t)(m0 + ar4[i]) * lda + k0 + ac4);
#pragma unroll
      for (int i = 0; i < 2; i++)
        rb[i] = *reinterpret_cast<const float4*>(B + (size_t)(k0 + bk2[i]) * ldb + n0 + bn4);
    }

    // Compute: preload all fragments, then issue 4 WMMAs
    v16h afrag = load_afrag(As, wave, lane);
    v16h bf0 = load_bfrag(Bs, 0, lane);
    v16h bf1 = load_bfrag(Bs, 1, lane);
    v16h bf2 = load_bfrag(Bs, 2, lane);
    v16h bf3 = load_bfrag(Bs, 3, lane);
    acc[0] = __builtin_amdgcn_wmma_f32_16x16x32_f16(false, afrag, false, bf0, (short)0, acc[0], false, false);
    acc[1] = __builtin_amdgcn_wmma_f32_16x16x32_f16(false, afrag, false, bf1, (short)0, acc[1], false, false);
    acc[2] = __builtin_amdgcn_wmma_f32_16x16x32_f16(false, afrag, false, bf2, (short)0, acc[2], false, false);
    acc[3] = __builtin_amdgcn_wmma_f32_16x16x32_f16(false, afrag, false, bf3, (short)0, acc[3], false, false);
    __syncthreads();
  }

  const int rbase = m0 + (wave << 4) + ((lane >> 4) << 3);
  const int cinl  = lane & 15;
#pragma unroll
  for (int t = 0; t < 4; t++) {
    int col = n0 + (t << 4) + cinl;
    float bb = bias[col];
#pragma unroll
    for (int j = 0; j < 8; j++) {
      C[(size_t)(rbase + j) * ldc + col] = act_apply<ACT>(acc[t][j] + bb);
    }
  }
}

// ---------------------------------------------------------------------------
// Guarded GEMM for ragged shapes (branch-free clamp+select guards).
// ---------------------------------------------------------------------------
template <int ACT>
__global__ __launch_bounds__(128) void gemm_wmma_kernel(
    const float* __restrict__ A, int lda,
    const float* __restrict__ B, int ldb,
    const float* __restrict__ bias,
    float* __restrict__ C, int ldc,
    int M, int N, int K)
{
  __shared__ _Float16 As[BM * AS_STRIDE];
  __shared__ _Float16 Bs[BN * BS_STRIDE];

  const int tid  = threadIdx.x;
  const int wave = tid >> 5;
  const int lane = tid & 31;
  const int m0 = blockIdx.y * BM;
  const int n0 = blockIdx.x * BN;

  v8f vzero = {};
  v8f acc[4];
#pragma unroll
  for (int t = 0; t < 4; t++) acc[t] = vzero;

  for (int k0 = 0; k0 < K; k0 += BK) {
#pragma unroll
    for (int i = 0; i < 16; i++) {
      int e = tid + i * 128;
      int r = e >> 5, c = e & 31;
      int gr = m0 + r, gc = k0 + c;
      int cr = min(gr, M - 1), cc = min(gc, K - 1);
      float v = A[(size_t)cr * lda + cc];
      v = (gr < M && gc < K) ? v : 0.0f;
      As[r * AS_STRIDE + c] = (_Float16)v;
    }
#pragma unroll
    for (int i = 0; i < 16; i++) {
      int e = tid + i * 128;
      int kk = e >> 6, n = e & 63;
      int gk = k0 + kk, gn = n0 + n;
      int ck = min(gk, K - 1), cn = min(gn, N - 1);
      float v = B[(size_t)ck * ldb + cn];
      v = (gk < K && gn < N) ? v : 0.0f;
      Bs[n * BS_STRIDE + kk] = (_Float16)v;
    }
    __syncthreads();

    v16h afrag = load_afrag(As, wave, lane);
    v16h bf0 = load_bfrag(Bs, 0, lane);
    v16h bf1 = load_bfrag(Bs, 1, lane);
    v16h bf2 = load_bfrag(Bs, 2, lane);
    v16h bf3 = load_bfrag(Bs, 3, lane);
    acc[0] = __builtin_amdgcn_wmma_f32_16x16x32_f16(false, afrag, false, bf0, (short)0, acc[0], false, false);
    acc[1] = __builtin_amdgcn_wmma_f32_16x16x32_f16(false, afrag, false, bf1, (short)0, acc[1], false, false);
    acc[2] = __builtin_amdgcn_wmma_f32_16x16x32_f16(false, afrag, false, bf2, (short)0, acc[2], false, false);
    acc[3] = __builtin_amdgcn_wmma_f32_16x16x32_f16(false, afrag, false, bf3, (short)0, acc[3], false, false);
    __syncthreads();
  }

  const int rbase = m0 + (wave << 4) + ((lane >> 4) << 3);
  const int cinl  = lane & 15;
#pragma unroll
  for (int t = 0; t < 4; t++) {
    int col = n0 + (t << 4) + cinl;
    if (col < N) {
      float bb = bias[col];
#pragma unroll
      for (int j = 0; j < 8; j++) {
        int row = rbase + j;
        if (row < M) {
          C[(size_t)row * ldc + col] = act_apply<ACT>(acc[t][j] + bb);
        }
      }
    }
  }
}

// ---------------------------------------------------------------------------
// LayerNorm over D=256 (wave per row): out = LN(x + res) * g + beta
// ---------------------------------------------------------------------------
__global__ __launch_bounds__(256) void ln_kernel(
    const float* __restrict__ x, const float* __restrict__ res,
    const float* __restrict__ g, const float* __restrict__ beta,
    float* __restrict__ out, int B)
{
  int wave = threadIdx.x >> 5, lane = threadIdx.x & 31;
  int row = blockIdx.x * 8 + wave;
  if (row >= B) return;
  size_t base = (size_t)row * DIM;
  float v[8];
  float s = 0.0f, s2 = 0.0f;
#pragma unroll
  for (int i = 0; i < 8; i++) {
    int d = lane + i * 32;
    float t = x[base + d] + (res ? res[base + d] : 0.0f);
    v[i] = t; s += t; s2 += t * t;
  }
#pragma unroll
  for (int off = 16; off > 0; off >>= 1) {
    s  += __shfl_xor(s,  off, 32);
    s2 += __shfl_xor(s2, off, 32);
  }
  float mean = s * (1.0f / DIM);
  float var  = s2 * (1.0f / DIM) - mean * mean;
  float inv  = rsqrtf(var + 1e-5f);
#pragma unroll
  for (int i = 0; i < 8; i++) {
    int d = lane + i * 32;
    out[base + d] = (v[i] - mean) * inv * g[d] + beta[d];
  }
}

__global__ void mul_kernel(const float* __restrict__ a, const float* __restrict__ b,
                           float* __restrict__ out, int n)
{
  int i = blockIdx.x * blockDim.x + threadIdx.x;
  if (i < n) out[i] = a[i] * b[i];
}

// ---------------------------------------------------------------------------
// SSM pointwise scan: block per row (256 threads = 256 channels)
// ---------------------------------------------------------------------------
__global__ __launch_bounds__(256) void ssm_kernel(
    const float* __restrict__ h, const float* __restrict__ delta,
    const float* __restrict__ Bm, const float* __restrict__ Cm,
    const float* __restrict__ A_log, const float* __restrict__ sg,
    float* __restrict__ y)
{
  int b = blockIdx.x, d = threadIdx.x;
  __shared__ float bc[16];
  if (d < 16) bc[d] = Bm[(size_t)b * 16 + d] * Cm[(size_t)b * 16 + d];
  __syncthreads();
  size_t idx = (size_t)b * DIM + d;
  float de = delta[idx];
  float acc = 0.0f;
#pragma unroll
  for (int s = 0; s < 16; s++) {
    float A = -expf(A_log[d * 16 + s]);
    acc += expf(de * A) * bc[s];
  }
  y[idx] = acc * de * h[idx] * sg[idx];
}

// ---------------------------------------------------------------------------
// Cross-attention over fixed KB (32 keys), H=4 heads of 64.
// ---------------------------------------------------------------------------
__global__ __launch_bounds__(128) void attn_kernel(
    const float* __restrict__ q, const float* __restrict__ kp,
    const float* __restrict__ vp, float* __restrict__ o)
{
  __shared__ float kl[KBR * DIM];
  __shared__ float vl[KBR * DIM];
  int tid = threadIdx.x;
#pragma unroll
  for (int i = 0; i < 16; i++) {
    int e4 = (tid + i * 128) << 2;
    *reinterpret_cast<float4*>(&kl[e4]) = *reinterpret_cast<const float4*>(&kp[e4]);
    *reinterpret_cast<float4*>(&vl[e4]) = *reinterpret_cast<const float4*>(&vp[e4]);
  }
  __syncthreads();

  int b = blockIdx.x * 32 + (tid >> 2);
  int head = tid & 3;
  int off = head * 64;
  const float* qr = q + (size_t)b * DIM + off;

  float sc[32];
  float mx = -1e30f;
#pragma unroll
  for (int n = 0; n < 32; n++) {
    float s = 0.0f;
    for (int d2 = 0; d2 < 64; d2++) s += qr[d2] * kl[n * DIM + off + d2];
    s *= 0.125f;
    sc[n] = s;
    mx = fmaxf(mx, s);
  }
  float sum = 0.0f;
#pragma unroll
  for (int n = 0; n < 32; n++) { sc[n] = expf(sc[n] - mx); sum += sc[n]; }
  float inv = 1.0f / sum;

  float* orow = o + (size_t)b * DIM + off;
  for (int d2 = 0; d2 < 64; d2++) {
    float a = 0.0f;
#pragma unroll
    for (int n = 0; n < 32; n++) a += sc[n] * vl[n * DIM + off + d2];
    orow[d2] = a * inv;
  }
}

// ---------------------------------------------------------------------------
// Top-2 gate
// ---------------------------------------------------------------------------
__global__ void moe_gate_kernel(const float* __restrict__ h,
                                const float* __restrict__ gw,
                                float* __restrict__ comb, int B)
{
  int b = blockIdx.x * blockDim.x + threadIdx.x;
  if (b >= B) return;
  float lg[NEXP];
#pragma unroll
  for (int e = 0; e < NEXP; e++) lg[e] = 0.0f;
  const float* hr = h + (size_t)b * DIM;
  for (int d = 0; d < DIM; d++) {
    float hv = hr[d];
#pragma unroll
    for (int e = 0; e < NEXP; e++) lg[e] += hv * gw[d * NEXP + e];
  }
  int bi = 0; float bv = lg[0];
#pragma unroll
  for (int e = 1; e < NEXP; e++) { if (lg[e] > bv) { bv = lg[e]; bi = e; } }
  int si = -1; float sv = -1e30f;
#pragma unroll
  for (int e = 0; e < NEXP; e++) { if (e != bi && lg[e] > sv) { sv = lg[e]; si = e; } }
  float m  = fmaxf(bv, sv);
  float e0 = expf(bv - m), e1 = expf(sv - m);
  float is = 1.0f / (e0 + e1);
  float w0 = e0 * is, w1 = e1 * is;
  float* cr = comb + (size_t)b * NEXP;
#pragma unroll
  for (int e = 0; e < NEXP; e++) cr[e] = (e == bi) ? w0 : ((e == si) ? w1 : 0.0f);
}

__global__ void scale_add_kernel(float* __restrict__ acc,
                                 const float* __restrict__ eo,
                                 const float* __restrict__ comb, int e, int n)
{
  int i = blockIdx.x * blockDim.x + threadIdx.x;
  if (i >= n) return;
  int b = i >> 8;
  acc[i] += comb[(size_t)b * NEXP + e] * eo[i];
}

// ---------------------------------------------------------------------------
// Host-side orchestration
// ---------------------------------------------------------------------------
static inline void gemm(const float* A, int lda, const float* B, int ldb,
                        const float* bias, float* C, int ldc,
                        int M, int N, int K, int act, hipStream_t s)
{
  bool fast = (M % BMF == 0) && (N % BN == 0) && (K % BK == 0) &&
              (lda % 4 == 0) && (ldb % 4 == 0);
  dim3 gridF(N / BN, M / BMF);
  dim3 gridG((N + BN - 1) / BN, (M + BM - 1) / BM);

#define GEMM_DISPATCH(ACTV)                                                        \
  do {                                                                             \
    if (fast)                                                                      \
      gemm_wmma_fast_kernel<ACTV><<<gridF, dim3(256), 0, s>>>(A, lda, B, ldb,      \
          bias, C, ldc, M, N, K);                                                  \
    else                                                                           \
      gemm_wmma_kernel<ACTV><<<gridG, dim3(128), 0, s>>>(A, lda, B, ldb,           \
          bias, C, ldc, M, N, K);                                                  \
  } while (0)

  switch (act) {
    case ACT_GELU:     GEMM_DISPATCH(ACT_GELU);     break;
    case ACT_SIGMOID:  GEMM_DISPATCH(ACT_SIGMOID);  break;
    case ACT_SILU:     GEMM_DISPATCH(ACT_SILU);     break;
    case ACT_SOFTPLUS: GEMM_DISPATCH(ACT_SOFTPLUS); break;
    default:           GEMM_DISPATCH(ACT_NONE);     break;
  }
#undef GEMM_DISPATCH
}

extern "C" void kernel_launch(void* const* d_in, const int* in_sizes, int n_in,
                              void* d_out, int out_size, void* d_ws, size_t ws_size,
                              hipStream_t stream)
{
  const int B = BATCH;
  auto P = [&](int i) { return (const float*)d_in[i]; };

  // Input order: x, then params dict in insertion order, then 3 blocks x 35.
  const float* x       = P(0);
  const float* ce_w    = P(1);  const float* ce_b    = P(2);
  const float* cat_w1  = P(3);  const float* cat_b1  = P(4);
  const float* cat_w2  = P(5);  const float* cat_b2  = P(6);
  const float* pg_w    = P(7);  const float* pg_b    = P(8);
  const float* ip_w    = P(9);  const float* ip_b    = P(10);
  const float* in_g    = P(11); const float* in_b    = P(12);
  const float* head_w1 = P(13); const float* head_b1 = P(14);
  const float* head_w2 = P(15); const float* head_b2 = P(16);
  const int BLK0 = 17, BSTRIDE = 35;

  // Workspace layout (floats)
  float* ws = (float*)d_ws;
  float* f_buf512 = ws;                                  // [B,512]
  float* f_h      = f_buf512 + (size_t)B * 512;          // [B,256]
  float* f_t0     = f_h      + (size_t)B * DIM;
  float* f_t1     = f_t0     + (size_t)B * DIM;
  float* f_t2     = f_t1     + (size_t)B * DIM;
  float* f_Bm     = f_t2     + (size_t)B * DIM;          // [B,16]
  float* f_Cm     = f_Bm     + (size_t)B * 16;
  float* f_comb   = f_Cm     + (size_t)B * 16;           // [B,8]
  float* f_kp     = f_comb   + (size_t)B * NEXP;         // [32,256]
  float* f_vp     = f_kp     + (size_t)KBR * DIM;

  const int nBD = B * DIM;
  const dim3 eb(256);
  const dim3 eg((nBD + 255) / 256);
  const dim3 lng(B / 8);

  // ---------------- Embedding ----------------
  gemm(x, 83, ce_w, DIM, ce_b, f_buf512, 512, B, DIM, 83, ACT_GELU, stream);
  gemm(x, 83, cat_w1, 128, cat_b1, f_t0, 128, B, 128, 83, ACT_GELU, stream);
  gemm(f_t0, 128, cat_w2, DIM, cat_b2, f_buf512 + 256, 512, B, DIM, 128, ACT_NONE, stream);
  gemm(f_buf512, 512, pg_w, DIM, pg_b, f_t0, DIM, B, DIM, 512, ACT_SIGMOID, stream);
  gemm(f_buf512, 512, ip_w, DIM, ip_b, f_t1, DIM, B, DIM, 512, ACT_NONE, stream);
  mul_kernel<<<eg, eb, 0, stream>>>(f_t1, f_t0, f_t2, nBD);
  ln_kernel<<<lng, eb, 0, stream>>>(f_t2, nullptr, in_g, in_b, f_h, B);

  // ---------------- Blocks ----------------
  for (int blk = 0; blk < 3; blk++) {
    int bi = BLK0 + blk * BSTRIDE;
    const float* dp_w = P(bi + 0);  const float* dp_b  = P(bi + 1);
    const float* Bp_w = P(bi + 2);  const float* Bp_b  = P(bi + 3);
    const float* Cp_w = P(bi + 4);  const float* Cp_b  = P(bi + 5);
    const float* A_lg = P(bi + 6);
    const float* sg_w = P(bi + 7);  const float* sg_b  = P(bi + 8);
    const float* so_w = P(bi + 9);  const float* so_b  = P(bi + 10);
    const float* ssm_g = P(bi + 11); const float* ssm_be = P(bi + 12);
    const float* kb   = P(bi + 13);
    const float* q_w  = P(bi + 14); const float* q_b   = P(bi + 15);
    const float* k_w  = P(bi + 16); const float* k_b   = P(bi + 17);
    const float* v_w  = P(bi + 18); const float* v_b   = P(bi + 19);
    const float* ao_w = P(bi + 20); const float* ao_b  = P(bi + 21);
    const float* at_g = P(bi + 22); const float* at_be = P(bi + 23);
    const float* gate_w = P(bi + 24);
    const float* e_w1 = P(bi + 25); const float* e_b1  = P(bi + 26);
    const float* e_w2 = P(bi + 27); const float* e_b2  = P(bi + 28);
    const float* s_w1 = P(bi + 29); const float* s_b1  = P(bi + 30);
    const float* s_w2 = P(bi + 31); const float* s_b2  = P(bi + 32);
    const float* moe_g = P(bi + 33); const float* moe_be = P(bi + 34);

    // --- SSM ---
    gemm(f_h, DIM, dp_w, DIM, dp_b, f_t0, DIM, B, DIM, DIM, ACT_SOFTPLUS, stream);
    gemm(f_h, DIM, Bp_w, 16, Bp_b, f_Bm, 16, B, 16, DIM, ACT_NONE, stream);
    gemm(f_h, DIM, Cp_w, 16, Cp_b, f_Cm, 16, B, 16, DIM, ACT_NONE, stream);
    gemm(f_h, DIM, sg_w, DIM, sg_b, f_t1, DIM, B, DIM, DIM, ACT_SILU, stream);
    ssm_kernel<<<dim3(B), dim3(DIM), 0, stream>>>(f_h, f_t0, f_Bm, f_Cm, A_lg, f_t1, f_t2);
    gemm(f_t2, DIM, so_w, DIM, so_b, f_t0, DIM, B, DIM, DIM, ACT_NONE, stream);
    ln_kernel<<<lng, eb, 0, stream>>>(f_t0, f_h, ssm_g, ssm_be, f_h, B);

    // --- Attention ---
    gemm(f_h, DIM, q_w, DIM, q_b, f_t0, DIM, B, DIM, DIM, ACT_NONE, stream);
    gemm(kb, DIM, k_w, DIM, k_b, f_kp, DIM, KBR, DIM, DIM, ACT_NONE, stream);
    gemm(kb, DIM, v_w, DIM, v_b, f_vp, DIM, KBR, DIM, DIM, ACT_NONE, stream);
    attn_kernel<<<dim3(B / 32), dim3(128), 0, stream>>>(f_t0, f_kp, f_vp, f_t1);
    gemm(f_t1, DIM, ao_w, DIM, ao_b, f_t2, DIM, B, DIM, DIM, ACT_NONE, stream);
    ln_kernel<<<lng, eb, 0, stream>>>(f_t2, f_h, at_g, at_be, f_h, B);

    // --- MoE ---
    moe_gate_kernel<<<dim3((B + 255) / 256), eb, 0, stream>>>(f_h, gate_w, f_comb, B);
    gemm(f_h, DIM, s_w1, FFD, s_b1, f_buf512, FFD, B, FFD, DIM, ACT_GELU, stream);
    gemm(f_buf512, FFD, s_w2, DIM, s_b2, f_t2, DIM, B, DIM, FFD, ACT_NONE, stream);
    for (int e = 0; e < NEXP; e++) {
      const float* w1 = e_w1 + (size_t)e * DIM * FFD;
      const float* b1 = e_b1 + (size_t)e * FFD;
      const float* w2 = e_w2 + (size_t)e * FFD * DIM;
      const float* b2 = e_b2 + (size_t)e * DIM;
      gemm(f_h, DIM, w1, FFD, b1, f_buf512, FFD, B, FFD, DIM, ACT_GELU, stream);
      gemm(f_buf512, FFD, w2, DIM, b2, f_t0, DIM, B, DIM, FFD, ACT_NONE, stream);
      scale_add_kernel<<<eg, eb, 0, stream>>>(f_t2, f_t0, f_comb, e, nBD);
    }
    ln_kernel<<<lng, eb, 0, stream>>>(f_t2, f_h, moe_g, moe_be, f_h, B);
  }

  // ---------------- Head ----------------
  gemm(f_h, DIM, head_w1, DIM, head_b1, f_t0, DIM, B, DIM, DIM, ACT_GELU, stream);
  gemm(f_t0, DIM, head_w2, NCLS, head_b2, (float*)d_out, NCLS, B, NCLS, DIM, ACT_NONE, stream);

  (void)in_sizes; (void)n_in; (void)out_size; (void)ws_size;
}
